// DocoderNoConcatEncoderDropConnect_50989851738694
// MI455X (gfx1250) — matrix operand order repaired and checked by reading, not compiled, not run on hardware
//
#include <hip/hip_runtime.h>

#define DEV __device__ __forceinline__

typedef __bf16 bf16;
typedef __attribute__((ext_vector_type(16))) __bf16 v16bf;
typedef __attribute__((ext_vector_type(8)))  __bf16 v8bf;
typedef __attribute__((ext_vector_type(8)))  float  v8f;

// Problem dims
#define BB   128
#define PP   196
#define FF   2048
#define HH   512
#define NFD  512
#define EE   512
#define VV   10000
#define LL   20
#define TT   19
#define KCAT 3072   // E + F + H   (xcat row)
#define NGATES 2048 // 4H
#define NHG  2560   // NF (ha) + F (gate)
#define EPSC 1e-5f

// ---------- scalar helpers ----------
DEV bf16 f2bf(float f) {
  union { float f; unsigned u; } a{f};
  unsigned u = a.u;
  u += 0x7FFFu + ((u >> 16) & 1u);            // round-to-nearest-even
  union { unsigned short s; bf16 b; } o{(unsigned short)(u >> 16)};
  return o.b;
}
DEV float bf2f(bf16 b) {
  union { bf16 b; unsigned short s; } a{b};
  union { unsigned u; float f; } o{(unsigned)a.s << 16};
  return o.f;
}
DEV float sigm(float x) { return 1.f / (1.f + __expf(-x)); }

// ---------- WMMA fragment loaders (ISA 7.12.2, wave32) ----------
// A: 16xK row-major tile. lane<16 -> M=lane, K-chunks {0..7, 16..23}; lane>=16 shifts K by 8.
DEV v16bf load_a(const bf16* X, int ld, int lane, int k0) {
  const bf16* p = X + (size_t)(lane & 15) * ld + (k0 + ((lane >> 4) << 3));
  v8bf lo = *(const v8bf*)p;
  v8bf hi = *(const v8bf*)(p + 16);
  v16bf a;
#pragma unroll
  for (int i = 0; i < 8; ++i) { a[i] = lo[i]; a[i + 8] = hi[i]; }
  return a;
}
// B = W^T: W is (N,K) row-major. lane -> N=lane&15; K-chunks {0..7, 16..23}; lane>=16 shifts K by 8.
DEV v16bf load_b(const bf16* W, int ld, int lane, int k0) {
  const bf16* p = W + (size_t)(lane & 15) * ld + (k0 + ((lane >> 4) << 3));
  v8bf lo = *(const v8bf*)p;
  v8bf hi = *(const v8bf*)(p + 16);
  v16bf b;
#pragma unroll
  for (int i = 0; i < 8; ++i) { b[i] = lo[i]; b[i + 8] = hi[i]; }
  return b;
}
DEV v8f wmma_bf(v16bf a, v16bf b, v8f c) {
  return __builtin_amdgcn_wmma_f32_16x16x32_bf16(false, a, false, b, (short)0, c, false, false);
}

// ---------- setup kernels ----------
__global__ void k_f2bf(const float* __restrict__ src, bf16* __restrict__ dst, int n) {
  int i = blockIdx.x * 256 + threadIdx.x;
  if (i < n) dst[i] = f2bf(src[i]);
}
// Whg = [Wha(512x512) ; Wfb(2048x512)] -> (2560x512) bf16
__global__ void k_build_whg(const float* __restrict__ Wha, const float* __restrict__ Wfb,
                            bf16* __restrict__ dst) {
  int i = blockIdx.x * 256 + threadIdx.x;
  if (i >= NHG * HH) return;
  int r = i / HH, c = i - r * HH;
  float v = (r < NFD) ? Wha[r * HH + c] : Wfb[(r - NFD) * HH + c];
  dst[i] = f2bf(v);
}
// Wcat = [Wih(2048x2560) | Whh(2048x512)] -> (2048x3072) bf16
__global__ void k_build_wcat(const float* __restrict__ Wih, const float* __restrict__ Whh,
                             bf16* __restrict__ dst) {
  int i = blockIdx.x * 256 + threadIdx.x;
  if (i >= NGATES * KCAT) return;
  int r = i / KCAT, c = i - r * KCAT;
  float v = (c < EE + FF) ? Wih[r * (EE + FF) + c] : Whh[r * HH + (c - (EE + FF))];
  dst[i] = f2bf(v);
}
// feat[b][f] = mean_p enc[b][p][f]
__global__ void k_feat_mean(const float* __restrict__ enc, float* __restrict__ feat) {
  int i = blockIdx.x * 256 + threadIdx.x;      // B*F threads
  if (i >= BB * FF) return;
  int b = i >> 11, f = i & (FF - 1);
  const float* p = enc + (size_t)b * PP * FF + f;
  float s = 0.f;
  for (int q = 0; q < PP; ++q) s += p[(size_t)q * FF];
  feat[i] = s * (1.f / (float)PP);
}
// h0/c0: wave-per-output reduction over K=2048, bn epilogue. h0 -> xcat h-slot (bf16), c0 -> c (f32)
__global__ void k_init_hc(const float* __restrict__ feat,
                          const float* __restrict__ Wh0, const float* __restrict__ bh0,
                          const float* __restrict__ Wc0, const float* __restrict__ bc0,
                          const float* __restrict__ gh, const float* __restrict__ bth,
                          const float* __restrict__ gc, const float* __restrict__ btc,
                          bf16* __restrict__ xcat, float* __restrict__ c) {
  int wave = threadIdx.x >> 5, lane = threadIdx.x & 31;
  int flat = blockIdx.x * 8 + wave;            // 0 .. 2*B*H-1
  int which = flat >> 16;                      // 0 = h, 1 = c   (B*H = 65536)
  int bj = flat & 65535;
  int b = bj >> 9, j = bj & (HH - 1);
  const float* W = which ? Wc0 : Wh0;
  const float* fr = feat + (size_t)b * FF;
  const float* wr = W + (size_t)j * FF;
  float s = 0.f;
  for (int i = lane; i < FF; i += 32) s += fr[i] * wr[i];
#pragma unroll
  for (int off = 16; off >= 1; off >>= 1) s += __shfl_xor(s, off, 32);
  if (lane == 0) {
    float rs = rsqrtf(1.f + EPSC);
    if (which) {
      float x = sigm(s + bc0[j]);
      c[b * HH + j] = gc[j] * (x * rs) + btc[j];
    } else {
      float x = sigm(s + bh0[j]);
      xcat[(size_t)b * KCAT + (EE + FF) + j] = f2bf(gh[j] * (x * rs) + bth[j]);
    }
  }
}

// ---------- feat_att = enc_bf(25088x2048) @ Wfa^T(->512) + bfa, stored bf16 ----------
// grid = 1568 M-tiles, block 256 (8 waves); wave computes 16x64 (4 N-tiles, A reuse x4)
// Software-pipelined: fragments for k+32 are loaded before the WMMAs for k issue.
__global__ void k_featatt(const bf16* __restrict__ enc, const bf16* __restrict__ Wfa,
                          const float* __restrict__ bfa, bf16* __restrict__ featt) {
  int wave = threadIdx.x >> 5, lane = threadIdx.x & 31;
  size_t m0 = (size_t)blockIdx.x * 16;
  int n0 = wave * 64;
  const bf16* A = enc + m0 * FF;
  const bf16* Bb[4];
#pragma unroll
  for (int j = 0; j < 4; ++j) Bb[j] = Wfa + (size_t)(n0 + j * 16) * FF;

  v8f acc[4] = {};
  v16bf a0 = load_a(A, FF, lane, 0);
  v16bf b0[4];
#pragma unroll
  for (int j = 0; j < 4; ++j) b0[j] = load_b(Bb[j], FF, lane, 0);

  for (int k0 = 0; k0 < FF - 32; k0 += 32) {
    int k1 = k0 + 32;
    __builtin_prefetch(A + (size_t)(lane & 15) * FF + k1 + 32, 0, 1);
    v16bf a1 = load_a(A, FF, lane, k1);
    v16bf b1[4];
#pragma unroll
    for (int j = 0; j < 4; ++j) b1[j] = load_b(Bb[j], FF, lane, k1);
#pragma unroll
    for (int j = 0; j < 4; ++j) acc[j] = wmma_bf(a0, b0[j], acc[j]);
    a0 = a1;
#pragma unroll
    for (int j = 0; j < 4; ++j) b0[j] = b1[j];
  }
#pragma unroll
  for (int j = 0; j < 4; ++j) acc[j] = wmma_bf(a0, b0[j], acc[j]);

  int half = lane >> 4, nl = lane & 15;
#pragma unroll
  for (int j = 0; j < 4; ++j) {
    int n = n0 + j * 16 + nl;
    float bias = bfa[n];
#pragma unroll
    for (int v = 0; v < 8; ++v) {
      size_t m = m0 + v + 8 * half;
      featt[m * NFD + n] = f2bf(acc[j][v] + bias);
    }
  }
}

// ---------- per-step S1: hg = h @ [Wha;Wfb]^T ; ha=+bha (f32), gate=sigmoid(+bfb) ----------
// 1280 wave-tiles (M=8 x N=160), grid=160 blocks x 8 waves, K=512, pipelined
__global__ void k_gemm_hg(const bf16* __restrict__ xcat, const bf16* __restrict__ Whg,
                          const float* __restrict__ bha, const float* __restrict__ bfb,
                          float* __restrict__ hg) {
  int wave = threadIdx.x >> 5, lane = threadIdx.x & 31;
  int flat = blockIdx.x * 8 + wave;
  int nt = flat % 160, mt = flat / 160;
  const bf16* A = xcat + (size_t)mt * 16 * KCAT + (EE + FF);   // h slot
  const bf16* Bp = Whg + (size_t)nt * 16 * HH;

  v8f acc = {};
  v16bf a0 = load_a(A, KCAT, lane, 0);
  v16bf b0 = load_b(Bp, HH, lane, 0);
  for (int k0 = 0; k0 < HH - 32; k0 += 32) {
    v16bf a1 = load_a(A, KCAT, lane, k0 + 32);
    v16bf b1 = load_b(Bp, HH, lane, k0 + 32);
    acc = wmma_bf(a0, b0, acc);
    a0 = a1; b0 = b1;
  }
  acc = wmma_bf(a0, b0, acc);

  int half = lane >> 4, n = nt * 16 + (lane & 15);
#pragma unroll
  for (int v = 0; v < 8; ++v) {
    int b_row = mt * 16 + v + 8 * half;
    float val = acc[v];
    hg[(size_t)b_row * NHG + n] =
        (n < NFD) ? (val + bha[n]) : sigm(val + bfb[n - NFD]);
  }
}

// ---------- S2: e[b][p] = sum_n relu(featt[bp][n] + ha[b][n]) * Wv[n] + bv ----------
// wave per (b,p); grid = B*25 blocks x 8 waves
__global__ void k_escore(const bf16* __restrict__ featt, const float* __restrict__ hg,
                         const float* __restrict__ Wv, const float* __restrict__ bv,
                         float* __restrict__ e) {
  int wave = threadIdx.x >> 5, lane = threadIdx.x & 31;
  int b = blockIdx.x / 25;
  int p = (blockIdx.x % 25) * 8 + wave;
  if (p >= PP) return;
  const bf16* row = featt + ((size_t)(b * PP + p)) * NFD;
  const float* ha = hg + (size_t)b * NHG;
  float s = 0.f;
#pragma unroll
  for (int i = 0; i < 16; ++i) {
    int n = i * 32 + lane;
    float v = bf2f(row[n]) + ha[n];
    s += fmaxf(v, 0.f) * Wv[n];
  }
#pragma unroll
  for (int off = 16; off >= 1; off >>= 1) s += __shfl_xor(s, off, 32);
  if (lane == 0) e[b * PP + p] = s + bv[0];
}

// ---------- S3: softmax over P, write alpha + masked alphas output ----------
__global__ void k_softmax(const float* __restrict__ e, const int* __restrict__ clen,
                          float* __restrict__ alpha, float* __restrict__ out_alpha, int t) {
  __shared__ float red[256];
  int b = blockIdx.x, tid = threadIdx.x;
  float v = (tid < PP) ? e[b * PP + tid] : -3.4e38f;
  red[tid] = v; __syncthreads();
  for (int s = 128; s > 0; s >>= 1) { if (tid < s) red[tid] = fmaxf(red[tid], red[tid + s]); __syncthreads(); }
  float mx = red[0]; __syncthreads();
  float ex = (tid < PP) ? __expf(v - mx) : 0.f;
  red[tid] = ex; __syncthreads();
  for (int s = 128; s > 0; s >>= 1) { if (tid < s) red[tid] += red[tid + s]; __syncthreads(); }
  float inv = 1.f / red[0];
  if (tid < PP) {
    float al = ex * inv;
    alpha[b * PP + tid] = al;
    float m = (t < clen[b] - 1) ? 1.f : 0.f;
    out_alpha[((size_t)b * TT + t) * PP + tid] = al * m;
  }
}

// ---------- S4: context + build xcat = [emb_t | gate*context | h] ----------
__global__ void k_context(const bf16* __restrict__ enc, const float* __restrict__ alpha,
                          const float* __restrict__ hg, const bf16* __restrict__ emb_bf,
                          const int* __restrict__ gt, bf16* __restrict__ xcat, int t) {
  int i = blockIdx.x * 256 + threadIdx.x;      // B*F
  if (i >= BB * FF) return;
  int b = i >> 11, f = i & (FF - 1);
  const bf16* eb = enc + (size_t)b * PP * FF + f;
  const float* al = alpha + b * PP;
  float s = 0.f;
  for (int p = 0; p < PP; ++p) s += bf2f(eb[(size_t)p * FF]) * al[p];
  float gc = hg[(size_t)b * NHG + NFD + f] * s;
  xcat[(size_t)b * KCAT + EE + f] = f2bf(gc);
  if (f < EE) {
    int id = gt[b * LL + t];
    xcat[(size_t)b * KCAT + f] = emb_bf[(size_t)id * EE + f];
  }
}

// ---------- S5: gates = xcat(128x3072) @ Wcat^T(2048x3072) ----------
// 256 waves (M=8 x Nstrip=32, strip = 16x64), grid = 32 blocks x 8 waves, K=3072, pipelined
__global__ void k_gemm_gates(const bf16* __restrict__ xcat, const bf16* __restrict__ Wcat,
                             float* __restrict__ gates) {
  int wave = threadIdx.x >> 5, lane = threadIdx.x & 31;
  int flat = blockIdx.x * 8 + wave;
  int mt = flat / 32, ns = flat % 32;
  const bf16* A = xcat + (size_t)mt * 16 * KCAT;
  const bf16* Bb[4];
#pragma unroll
  for (int j = 0; j < 4; ++j) Bb[j] = Wcat + (size_t)(ns * 64 + j * 16) * KCAT;

  v8f acc[4] = {};
  v16bf a0 = load_a(A, KCAT, lane, 0);
  v16bf b0[4];
#pragma unroll
  for (int j = 0; j < 4; ++j) b0[j] = load_b(Bb[j], KCAT, lane, 0);

  for (int k0 = 0; k0 < KCAT - 32; k0 += 32) {
    int k1 = k0 + 32;
    __builtin_prefetch(A + (size_t)(lane & 15) * KCAT + k1 + 32, 0, 1);
    v16bf a1 = load_a(A, KCAT, lane, k1);
    v16bf b1[4];
#pragma unroll
    for (int j = 0; j < 4; ++j) b1[j] = load_b(Bb[j], KCAT, lane, k1);
#pragma unroll
    for (int j = 0; j < 4; ++j) acc[j] = wmma_bf(a0, b0[j], acc[j]);
    a0 = a1;
#pragma unroll
    for (int j = 0; j < 4; ++j) b0[j] = b1[j];
  }
#pragma unroll
  for (int j = 0; j < 4; ++j) acc[j] = wmma_bf(a0, b0[j], acc[j]);

  int half = lane >> 4, nl = lane & 15;
#pragma unroll
  for (int j = 0; j < 4; ++j) {
    int n = ns * 64 + j * 16 + nl;
#pragma unroll
    for (int v = 0; v < 8; ++v) {
      int b_row = mt * 16 + v + 8 * half;
      gates[(size_t)b_row * NGATES + n] = acc[j][v];
    }
  }
}

// ---------- S6: LSTM pointwise; h_new -> xcat h-slot (bf16), c updated in place ----------
__global__ void k_lstm(const float* __restrict__ gates,
                       const float* __restrict__ bih, const float* __restrict__ bhh,
                       float* __restrict__ c, bf16* __restrict__ xcat) {
  int i = blockIdx.x * 256 + threadIdx.x;      // B*H
  if (i >= BB * HH) return;
  int b = i >> 9, j = i & (HH - 1);
  const float* g = gates + (size_t)b * NGATES;
  float xi = g[j]            + bih[j]            + bhh[j];
  float xf = g[HH + j]       + bih[HH + j]       + bhh[HH + j];
  float xg = g[2 * HH + j]   + bih[2 * HH + j]   + bhh[2 * HH + j];
  float xo = g[3 * HH + j]   + bih[3 * HH + j]   + bhh[3 * HH + j];
  float cn = sigm(xf) * c[i] + sigm(xi) * tanhf(xg);
  float hn = sigm(xo) * tanhf(cn);
  c[i] = cn;
  xcat[(size_t)b * KCAT + (EE + FF) + j] = f2bf(hn);
}

// ---------- S7: preds = h_new @ Wfc^T(10000x512) + bfc, masked store to d_out ----------
// 5000 wave-tiles (M=8 x N=625), grid = 625 blocks x 8 waves, K=512, pipelined
__global__ void k_gemm_preds(const bf16* __restrict__ xcat, const bf16* __restrict__ Wfc,
                             const float* __restrict__ bfc, const int* __restrict__ clen,
                             float* __restrict__ out, int t) {
  int wave = threadIdx.x >> 5, lane = threadIdx.x & 31;
  int flat = blockIdx.x * 8 + wave;
  int nt = flat % 625, mt = flat / 625;
  const bf16* A = xcat + (size_t)mt * 16 * KCAT + (EE + FF);
  const bf16* Bp = Wfc + (size_t)nt * 16 * HH;

  v8f acc = {};
  v16bf a0 = load_a(A, KCAT, lane, 0);
  v16bf b0 = load_b(Bp, HH, lane, 0);
  for (int k0 = 0; k0 < HH - 32; k0 += 32) {
    int k1 = k0 + 32;
    __builtin_prefetch(Bp + (size_t)(lane & 15) * HH + k1 + 32, 0, 1);
    v16bf a1 = load_a(A, KCAT, lane, k1);
    v16bf b1 = load_b(Bp, HH, lane, k1);
    acc = wmma_bf(a0, b0, acc);
    a0 = a1; b0 = b1;
  }
  acc = wmma_bf(a0, b0, acc);

  int half = lane >> 4, n = nt * 16 + (lane & 15);
  float bias = bfc[n];
#pragma unroll
  for (int v = 0; v < 8; ++v) {
    int b_row = mt * 16 + v + 8 * half;
    float m = (t < clen[b_row] - 1) ? 1.f : 0.f;
    out[((size_t)b_row * TT + t) * VV + n] = m * (acc[v] + bias);
  }
}

// ---------- host ----------
extern "C" void kernel_launch(void* const* d_in, const int* in_sizes, int n_in,
                              void* d_out, int out_size, void* d_ws, size_t ws_size,
                              hipStream_t stream) {
  (void)in_sizes; (void)n_in; (void)out_size; (void)ws_size;
  const float* enc   = (const float*)d_in[0];
  const float* emb   = (const float*)d_in[1];
  const float* Wfa   = (const float*)d_in[2];
  const float* bfa   = (const float*)d_in[3];
  const float* Wha   = (const float*)d_in[4];
  const float* bha   = (const float*)d_in[5];
  const float* Wv    = (const float*)d_in[6];
  const float* bv    = (const float*)d_in[7];
  const float* Wh0   = (const float*)d_in[8];
  const float* bh0   = (const float*)d_in[9];
  const float* Wc0   = (const float*)d_in[10];
  const float* bc0   = (const float*)d_in[11];
  const float* gh    = (const float*)d_in[12];
  const float* bth   = (const float*)d_in[13];
  const float* gc    = (const float*)d_in[14];
  const float* btc   = (const float*)d_in[15];
  const float* Wfb   = (const float*)d_in[16];
  const float* bfb   = (const float*)d_in[17];
  const float* Wih   = (const float*)d_in[18];
  const float* Whh   = (const float*)d_in[19];
  const float* bih   = (const float*)d_in[20];
  const float* bhh   = (const float*)d_in[21];
  const float* Wfc   = (const float*)d_in[22];
  const float* bfc   = (const float*)d_in[23];
  const int*   gt    = (const int*)d_in[24];
  const int*   clen  = (const int*)d_in[25];

  float* out_pred  = (float*)d_out;                       // (B,T,V)
  float* out_alpha = out_pred + (size_t)BB * TT * VV;     // (B,T,P)

  // workspace carve (256B aligned)
  char* wp = (char*)d_ws;
  auto take = [&](size_t bytes) -> void* {
    void* r = (void*)wp;
    wp += (bytes + 255) & ~(size_t)255;
    return r;
  };
  bf16* enc_bf  = (bf16*)take((size_t)BB * PP * FF * 2);
  bf16* featt   = (bf16*)take((size_t)BB * PP * NFD * 2);
  bf16* Wfa_bf  = (bf16*)take((size_t)NFD * FF * 2);
  bf16* Whg_bf  = (bf16*)take((size_t)NHG * HH * 2);
  bf16* Wcat_bf = (bf16*)take((size_t)NGATES * KCAT * 2);
  bf16* Wfc_bf  = (bf16*)take((size_t)VV * HH * 2);
  bf16* emb_bf  = (bf16*)take((size_t)VV * EE * 2);
  bf16* xcat    = (bf16*)take((size_t)BB * KCAT * 2);
  float* feat   = (float*)take((size_t)BB * FF * 4);
  float* hg     = (float*)take((size_t)BB * NHG * 4);
  float* ebuf   = (float*)take((size_t)BB * PP * 4);
  float* alpha  = (float*)take((size_t)BB * PP * 4);
  float* gates  = (float*)take((size_t)BB * NGATES * 4);
  float* cbuf   = (float*)take((size_t)BB * HH * 4);

  const int TPB = 256;
  auto nb = [](size_t n) { return (unsigned)((n + 255) / 256); };

  // one-time setup
  k_f2bf<<<nb((size_t)BB * PP * FF), TPB, 0, stream>>>(enc, enc_bf, BB * PP * FF);
  k_f2bf<<<nb((size_t)NFD * FF), TPB, 0, stream>>>(Wfa, Wfa_bf, NFD * FF);
  k_f2bf<<<nb((size_t)VV * HH), TPB, 0, stream>>>(Wfc, Wfc_bf, VV * HH);
  k_f2bf<<<nb((size_t)VV * EE), TPB, 0, stream>>>(emb, emb_bf, VV * EE);
  k_build_whg<<<nb((size_t)NHG * HH), TPB, 0, stream>>>(Wha, Wfb, Whg_bf);
  k_build_wcat<<<nb((size_t)NGATES * KCAT), TPB, 0, stream>>>(Wih, Whh, Wcat_bf);
  k_feat_mean<<<nb((size_t)BB * FF), TPB, 0, stream>>>(enc, feat);
  k_init_hc<<<(2 * BB * HH) / 8, TPB, 0, stream>>>(feat, Wh0, bh0, Wc0, bc0,
                                                   gh, bth, gc, btc, xcat, cbuf);
  k_featatt<<<(BB * PP) / 16, TPB, 0, stream>>>(enc_bf, Wfa_bf, bfa, featt);

  // recurrence
  for (int t = 0; t < TT; ++t) {
    k_gemm_hg<<<160, TPB, 0, stream>>>(xcat, Whg_bf, bha, bfb, hg);
    k_escore<<<BB * 25, TPB, 0, stream>>>(featt, hg, Wv, bv, ebuf);
    k_softmax<<<BB, TPB, 0, stream>>>(ebuf, clen, alpha, out_alpha, t);
    k_context<<<nb((size_t)BB * FF), TPB, 0, stream>>>(enc_bf, alpha, hg, emb_bf, gt, xcat, t);
    k_gemm_gates<<<32, TPB, 0, stream>>>(xcat, Wcat_bf, gates);
    k_lstm<<<nb((size_t)BB * HH), TPB, 0, stream>>>(gates, bih, bhh, cbuf, xcat);
    k_gemm_preds<<<625, TPB, 0, stream>>>(xcat, Wfc_bf, bfc, clen, out_pred, t);
  }
}
